// VPSLModel_81896436400670
// MI455X (gfx1250) — matrix operand order repaired and checked by reading, not compiled, not run on hardware
//
#include <hip/hip_runtime.h>
#include <math.h>

// N=20000 nodes, L=64, H1=H2=128.
// Per node: h = relu(S@W1+b1) [64x128]; logits = h@W2+b2 [64x64];
// masked softmax; gather via back-pointers; y=relu(mid@W3+b3); out=sigmoid(y@W4+b4).
// Memory-bound (665MB stream vs 42 GFLOP @ 23.3 TB/s => ~28us floor).
// One block (4 wave32) per node; weights staged as f16 in LDS; both GEMMs via
// v_wmma_f32_16x16x32_f16; mask prefetched via global_prefetch_b8 so the
// late mask reads hit in cache.

typedef __attribute__((ext_vector_type(16))) _Float16 v16h;
typedef __attribute__((ext_vector_type(8)))  float    v8f;

#define TL 64      // L (max_len / input dim)
#define TH 128     // H1 == H2

// padded LDS strides (in halves / floats) for bank-conflict avoidance
#define LDA_S  68   // S   rows:  [64][68]  f16
#define LDW1T  68   // W1T rows:  [128][68] f16  (k along row)
#define LDW2T  132  // W2T rows:  [64][132] f16  (k along row)
#define LDH    132  // h   rows:  [64][132] f16
#define LDLOG  65   // logits:    [64][65]  f32

// byte offsets inside one shared blob (total 60672 B < 64KB)
#define OFF_A    0        // 64*68*2   = 8704
#define OFF_W1T  8704     // 128*68*2  = 17408  -> ends 26112
#define OFF_W2T  26112    // 64*132*2  = 16896  -> ends 43008
#define OFF_H    43008    // 64*132*2  = 16896  -> ends 59904
#define OFF_LOG  0        // 64*65*4   = 16640  (overlays A+W1T after barrier)
#define OFF_MID  59904    // 64*4      = 256
#define OFF_RED  60160    // 128*4     = 512    -> ends 60672
#define SMEM_BYTES 60672

__global__ __launch_bounds__(128)
void vpsl_node_kernel(const float* __restrict__ S,    // [N,64,64]
                      const int*   __restrict__ mask, // [N,64,64]
                      const int*   __restrict__ gidx, // [N,64]
                      const int*   __restrict__ rowv, // [N,64]
                      const float* __restrict__ W1, const float* __restrict__ b1,
                      const float* __restrict__ W2, const float* __restrict__ b2,
                      const float* __restrict__ W3, const float* __restrict__ b3,
                      const float* __restrict__ W4, const float* __restrict__ b4,
                      float* __restrict__ out, int N)
{
    __shared__ __align__(16) char smem[SMEM_BYTES];
    _Float16* sA   = (_Float16*)(smem + OFF_A);
    _Float16* sW1T = (_Float16*)(smem + OFF_W1T);
    _Float16* sW2T = (_Float16*)(smem + OFF_W2T);
    _Float16* sH   = (_Float16*)(smem + OFF_H);
    float*    sLog = (float*)   (smem + OFF_LOG);
    float*    sMid = (float*)   (smem + OFF_MID);
    float*    sRed = (float*)   (smem + OFF_RED);

    const int n = blockIdx.x;
    if (n >= N) return;
    const int tid  = threadIdx.x;
    const int lane = tid & 31;
    const int wave = tid >> 5;          // 0..3
    const int lh   = lane >> 4;         // half-wave: 0/1
    const int ll   = lane & 15;
    const int m0   = wave * 16;         // 16-row strip per wave

    // ---- prefetch the mask (consumed last, half the HBM traffic):
    //      128 threads x one 128B line = whole 16KB mask; overlaps both GEMMs.
    const int* mrow = mask + (size_t)n * (TL * TL);
    __builtin_prefetch(mrow + tid * 32, 0, 1);           // global_prefetch_b8
    if (tid == 0) {
        __builtin_prefetch(gidx + (size_t)n * TL, 0, 1);
        __builtin_prefetch(rowv + (size_t)n * TL, 0, 1);
    }

    // ---- stage: S (f32->f16) and W1/W2 (transposed f16) into LDS ----
    const float* Sn = S + (size_t)n * (TL * TL);
    for (int i = tid; i < (TL * TL) / 4; i += 128) {
        float4 f = ((const float4*)Sn)[i];
        int r = (i * 4) / TL, c = (i * 4) % TL;
        _Float16* p = &sA[r * LDA_S + c];
        p[0] = (_Float16)f.x; p[1] = (_Float16)f.y;
        p[2] = (_Float16)f.z; p[3] = (_Float16)f.w;
    }
    for (int i = tid; i < TL * TH; i += 128) {        // W1 [64][128] -> W1T[h][k]
        int k = i / TH, h = i % TH;
        sW1T[h * LDW1T + k] = (_Float16)W1[i];
    }
    for (int i = tid; i < TH * TL; i += 128) {        // W2 [128][64] -> W2T[c][k]
        int k = i / TL, c = i % TL;
        sW2T[c * LDW2T + k] = (_Float16)W2[i];
    }
    __syncthreads();

    // ---- GEMM1: h = relu(S @ W1 + b1), 64x128, K=64 ----
    v8f acc[8];
    #pragma unroll
    for (int t = 0; t < 8; ++t)
        #pragma unroll
        for (int e = 0; e < 8; ++e) acc[t][e] = 0.0f;

    #pragma unroll
    for (int kk = 0; kk < TL; kk += 32) {
        union { v16h v; unsigned u[8]; } af;
        const int arow = m0 + ll;
        #pragma unroll
        for (int v = 0; v < 8; ++v) {
            int k = kk + (v >> 2) * 16 + lh * 8 + (v & 3) * 2;   // A f16 layout
            af.u[v] = *(const unsigned*)&sA[arow * LDA_S + k];
        }
        #pragma unroll
        for (int t = 0; t < 8; ++t) {
            union { v16h v; unsigned u[8]; } bf;
            const int bn = t * 16 + ll;
            #pragma unroll
            for (int v = 0; v < 8; ++v) {
                int k = kk + (v >> 2) * 16 + lh * 8 + (v & 3) * 2; // B f16 layout
                bf.u[v] = *(const unsigned*)&sW1T[bn * LDW1T + k];
            }
            acc[t] = __builtin_amdgcn_wmma_f32_16x16x32_f16(
                false, af.v, false, bf.v, (short)0, acc[t], false, false);
        }
    }
    // bias + relu -> sH (f16)
    #pragma unroll
    for (int t = 0; t < 8; ++t) {
        const int ncol = t * 16 + ll;
        const float bb = b1[ncol];
        #pragma unroll
        for (int v = 0; v < 8; ++v) {
            const int m = m0 + lh * 8 + v;
            float val = acc[t][v] + bb;
            sH[m * LDH + ncol] = (_Float16)(val > 0.0f ? val : 0.0f);
        }
    }
    __syncthreads();   // all waves done with sA/sW1T -> sLog may overlay them

    // ---- GEMM2: logits = h @ W2 + b2, 64x64, K=128 ----
    v8f acc2[4];
    #pragma unroll
    for (int t = 0; t < 4; ++t)
        #pragma unroll
        for (int e = 0; e < 8; ++e) acc2[t][e] = 0.0f;

    #pragma unroll
    for (int kk = 0; kk < TH; kk += 32) {
        union { v16h v; unsigned u[8]; } af;
        const int arow = m0 + ll;
        #pragma unroll
        for (int v = 0; v < 8; ++v) {
            int k = kk + (v >> 2) * 16 + lh * 8 + (v & 3) * 2;
            af.u[v] = *(const unsigned*)&sH[arow * LDH + k];
        }
        #pragma unroll
        for (int t = 0; t < 4; ++t) {
            union { v16h v; unsigned u[8]; } bf;
            const int bn = t * 16 + ll;
            #pragma unroll
            for (int v = 0; v < 8; ++v) {
                int k = kk + (v >> 2) * 16 + lh * 8 + (v & 3) * 2;
                bf.u[v] = *(const unsigned*)&sW2T[bn * LDW2T + k];
            }
            acc2[t] = __builtin_amdgcn_wmma_f32_16x16x32_f16(
                false, af.v, false, bf.v, (short)0, acc2[t], false, false);
        }
    }
    // bias + mask -> sLog (the single required read of sample_mask; prefetched)
    #pragma unroll
    for (int t = 0; t < 4; ++t) {
        const int ncol = t * 16 + ll;
        const float bb = b2[ncol];
        #pragma unroll
        for (int v = 0; v < 8; ++v) {
            const int m = m0 + lh * 8 + v;
            const float lg = acc2[t][v] + bb;
            sLog[m * LDLOG + ncol] = mrow[m * TL + ncol] ? lg : -INFINITY;
        }
    }
    __syncthreads();

    // ---- masked softmax + back-pointer gather: mid[j] ----
    if (tid < TL) {
        const int j = tid;
        const float* row = &sLog[j * LDLOG];
        float mx = -INFINITY;
        #pragma unroll 8
        for (int c = 0; c < TL; ++c) mx = fmaxf(mx, row[c]);
        float mid = 0.0f;
        if (mx > -INFINITY && rowv[(size_t)n * TL + j] == 1) {
            float s = 0.0f;
            #pragma unroll 8
            for (int c = 0; c < TL; ++c) s += __expf(row[c] - mx);  // exp(-inf)=0
            const int g = gidx[(size_t)n * TL + j];
            const float lg = row[g];
            mid = (lg == -INFINITY) ? 0.0f : __expf(lg - mx) / s;
        }
        sMid[j] = mid;
    }
    __syncthreads();

    // ---- final MLP: y = relu(mid @ W3 + b3); out = sigmoid(y . W4 + b4) ----
    {
        const int h = tid;            // 0..127
        float a3 = b3[h];
        #pragma unroll 8
        for (int j = 0; j < TL; ++j)
            a3 = fmaf(sMid[j], W3[j * TH + h], a3);
        const float y = a3 > 0.0f ? a3 : 0.0f;
        sRed[tid] = y * W4[h];
    }
    __syncthreads();
    #pragma unroll
    for (int s = 64; s > 0; s >>= 1) {
        if (tid < s) sRed[tid] += sRed[tid + s];
        __syncthreads();
    }
    if (tid == 0) {
        const float z = sRed[0] + b4[0];
        out[n] = 1.0f / (1.0f + __expf(-z));
    }
}

extern "C" void kernel_launch(void* const* d_in, const int* in_sizes, int n_in,
                              void* d_out, int out_size, void* d_ws, size_t ws_size,
                              hipStream_t stream) {
    const float* S    = (const float*)d_in[0];
    const int*   mask = (const int*)  d_in[1];
    const int*   gidx = (const int*)  d_in[2];
    const int*   rowv = (const int*)  d_in[3];
    const float* W1   = (const float*)d_in[4];
    const float* b1   = (const float*)d_in[5];
    const float* W2   = (const float*)d_in[6];
    const float* b2   = (const float*)d_in[7];
    const float* W3   = (const float*)d_in[8];
    const float* b3   = (const float*)d_in[9];
    const float* W4   = (const float*)d_in[10];
    const float* b4   = (const float*)d_in[11];
    float* out = (float*)d_out;

    const int N = in_sizes[0] / (TL * TL);   // 20000
    vpsl_node_kernel<<<N, 128, 0, stream>>>(S, mask, gidx, rowv,
                                            W1, b1, W2, b2, W3, b3, W4, b4,
                                            out, N);
}
